// RAFTGMA_75239237092023
// MI455X (gfx1250) — matrix operand order repaired and checked by reading, not compiled, not run on hardware
//
#include <hip/hip_runtime.h>
#include <hip/hip_bf16.h>
#include <stdint.h>

// Problem constants (from reference): B=2, C=256, H=64, W=128, N=8192, k=32
#define B_    2
#define C_    256
#define H_    64
#define W_    128
#define N_    (H_ * W_)       // 8192
#define TOPK  32
#define NT_   (N_ / 16)       // 512 16-wide tiles along N
#define KB_   (C_ / 32)       // 8 K-blocks of 32 for bf16 WMMA
#define TILE_DW (KB_ * 32 * 8) // 2048 dwords = one packed 16-col B tile (8 KB)
#define NEG_INF (-3.0e38f)

typedef __attribute__((ext_vector_type(16))) __bf16 v16bf;
typedef __attribute__((ext_vector_type(8)))  float  v8f;
typedef unsigned int u32x4 __attribute__((ext_vector_type(4)));
typedef int          i32x8 __attribute__((ext_vector_type(8)));
typedef int          i32x4 __attribute__((ext_vector_type(4)));

__device__ __forceinline__ unsigned short bf16_rne(float f) {
  unsigned u = __float_as_uint(f);
  return (unsigned short)((u + 0x7FFFu + ((u >> 16) & 1u)) >> 16);
}
__device__ __forceinline__ float bf16_up(unsigned short h) {
  return __uint_as_float(((unsigned)h) << 16);
}
__device__ __forceinline__ float lanebcast_f(float v, int srclane) {
  return __int_as_float(__builtin_amdgcn_ds_bpermute(srclane << 2, __float_as_int(v)));
}
__device__ __forceinline__ int lanebcast_i(int v, int srclane) {
  return __builtin_amdgcn_ds_bpermute(srclane << 2, v);
}

// TDM 1-D copy: ndw dwords from global to LDS (D# per CDNA5 ISA §8.3/8.4).
// This toolchain exposes the 6-arg builtin: (g0, g1, g2, g3, g4, cpol).
__device__ __forceinline__ void tdm_load_1d(uint32_t lds_byte_off,
                                            const uint32_t* gsrc,
                                            uint32_t ndw) {
  uint64_t ga = (uint64_t)(uintptr_t)gsrc;
  u32x4 g0;
  g0.x = 1u;                                            // count=1 (valid user D#)
  g0.y = lds_byte_off;                                  // lds_addr[31:0]
  g0.z = (uint32_t)ga;                                  // global_addr[31:0]
  g0.w = (uint32_t)((ga >> 32) & 0x01FFFFFFu) | (2u << 30); // addr[56:32] | type=2
  i32x8 g1;
  g1[0] = 0x00020000;                                   // data_size=4B; no mcast/pad
  g1[1] = (int)((ndw & 0xFFFFu) << 16);                 // tensor_dim0[15:0]
  g1[2] = (int)(((ndw >> 16) & 0xFFFFu) | (1u << 16));  // dim0[31:16] | tensor_dim1=1
  g1[3] = (int)((ndw & 0xFFFFu) << 16);                 // tile_dim0 = ndw
  g1[4] = 1;                                            // tile_dim1=1, tile_dim2=0
  g1[5] = (int)ndw;                                     // tensor_dim0_stride
  g1[6] = 0;
  g1[7] = 0;
  i32x4 z4 = {0, 0, 0, 0};                              // groups 2/3 unused (<=2D)
  i32x8 z8 = {0, 0, 0, 0, 0, 0, 0, 0};
  __builtin_amdgcn_tensor_load_to_lds(g0, g1, z4, z4, z8, 0);
}

// ---------------------------------------------------------------------------
// Pack kernel: f32 [B,C,N] -> split-bf16 (hi/lo) tiles in WMMA-native,
// per-lane-contiguous layout (A: 16x32 bf16 A-layout from f1; B: 32x16
// bf16 B-layout from f2).
// ---------------------------------------------------------------------------
__global__ void RAFTGMA_pack(const float* __restrict__ f1,
                             const float* __restrict__ f2,
                             uint32_t* __restrict__ pAhi, uint32_t* __restrict__ pAlo,
                             uint32_t* __restrict__ pBhi, uint32_t* __restrict__ pBlo) {
  int tid  = blockIdx.x * blockDim.x + threadIdx.x;
  int lane = tid & 31;
  int kb   = (tid >> 5) & (KB_ - 1);
  int t    = (tid >> 8) & (NT_ - 1);
  int b    = tid >> 17;
  int li   = lane & 15, h = lane >> 4;
  size_t base = ((((size_t)b * NT_ + t) * KB_ + kb) * 32 + lane) * 8;

  { // A operand from f1: lane li holds row n=t*16+li
    int n = t * 16 + li;
    const float* f = f1 + (size_t)b * C_ * N_ + n;
#pragma unroll
    for (int j = 0; j < 8; ++j) {
      int kk = ((j >> 2) << 4) + h * 8 + ((j & 3) << 1);
      int c  = kb * 32 + kk;
      float a0 = f[(size_t)c * N_];
      float a1 = f[(size_t)(c + 1) * N_];
      unsigned short h0 = bf16_rne(a0), h1 = bf16_rne(a1);
      unsigned short l0 = bf16_rne(a0 - bf16_up(h0));
      unsigned short l1 = bf16_rne(a1 - bf16_up(h1));
      pAhi[base + j] = ((unsigned)h1 << 16) | h0;
      pAlo[base + j] = ((unsigned)l1 << 16) | l0;
    }
  }
  { // B operand from f2: lane li holds column m=t*16+li
    int m = t * 16 + li;
    const float* f = f2 + (size_t)b * C_ * N_ + m;
#pragma unroll
    for (int j = 0; j < 8; ++j) {
      int k = kb * 32 + ((j >> 2) << 4) + h * 8 + ((j & 3) << 1);
      float b0 = f[(size_t)k * N_];
      float b1 = f[(size_t)(k + 1) * N_];
      unsigned short h0 = bf16_rne(b0), h1 = bf16_rne(b1);
      unsigned short l0 = bf16_rne(b0 - bf16_up(h0));
      unsigned short l1 = bf16_rne(b1 - bf16_up(h1));
      pBhi[base + j] = ((unsigned)h1 << 16) | h0;
      pBlo[base + j] = ((unsigned)l1 << 16) | l0;
    }
  }
}

// ---------------------------------------------------------------------------
// Fused split-bf16 WMMA GEMM + streaming top-32 + exact f32 rescore.
// 128 rows per workgroup (one 16-row tile per wave); all 8 waves march over
// the same column tile, which is TDM double-buffered into LDS (8x B reuse vs
// per-wave streaming -> B traffic drops to 512 MB/batch, L2-resident).
// Each wave's top-32 list is final for its rows (covers all columns).
// ---------------------------------------------------------------------------
__global__ __launch_bounds__(256)
void RAFTGMA_main(const float* __restrict__ f1, const float* __restrict__ f2,
                  const uint32_t* __restrict__ pAhi, const uint32_t* __restrict__ pAlo,
                  const uint32_t* __restrict__ pBhi, const uint32_t* __restrict__ pBlo,
                  float* __restrict__ out) {
  __shared__ __attribute__((aligned(32))) uint32_t s_bhi[2][TILE_DW]; // 2 x 8 KB
  __shared__ __attribute__((aligned(32))) uint32_t s_blo[2][TILE_DW]; // 2 x 8 KB
  __shared__ float s_vals[8 * 16 * TOPK];   // [wave][row][slot]
  __shared__ int   s_idxs[8 * 16 * TOPK];
  __shared__ float s_minv[8 * 16];
  __shared__ int   s_mins[8 * 16];

  int b    = blockIdx.x >> 6;               // 64 workgroups per batch
  int wave = threadIdx.x >> 5;
  int lane = threadIdx.x & 31;
  int li   = lane & 15, half = lane >> 4;
  int t    = ((blockIdx.x & 63) << 3) + wave;   // this wave's 16-row tile

  for (int i = threadIdx.x; i < 8 * 16 * TOPK; i += 256) {
    s_vals[i] = NEG_INF;
    s_idxs[i] = 0x7fffffff;
  }
  if (threadIdx.x < 8 * 16) { s_minv[threadIdx.x] = NEG_INF; s_mins[threadIdx.x] = 0; }

  // A fragments (hi+lo) for this wave's 16 rows stay resident in VGPRs.
  size_t abase = (((size_t)b * NT_ + t) * KB_) * 32 * 8;
  v16bf ahi[KB_], alo[KB_];
#pragma unroll
  for (int kb = 0; kb < KB_; ++kb) {
    size_t o = abase + ((size_t)kb * 32 + lane) * 8;
    ahi[kb] = *(const v16bf*)(pAhi + o);
    alo[kb] = *(const v16bf*)(pAlo + o);
  }

  // Prologue: TDM-load column tile u=0 into buffer 0.
  if (wave == 0) {
    size_t o0 = (((size_t)b * NT_) * KB_) * 32 * 8;
    tdm_load_1d((uint32_t)(uintptr_t)&s_bhi[0][0], pBhi + o0, TILE_DW);
    tdm_load_1d((uint32_t)(uintptr_t)&s_blo[0][0], pBlo + o0, TILE_DW);
  }
  __syncthreads();   // also covers the top-k LDS init

  for (int u = 0; u < NT_; ++u) {
    int cur = u & 1, nxt = cur ^ 1;
    if (wave == 0) {
      if (u + 1 < NT_) {
        // Prefetch tile u+1 (buffer was released by the end-of-iter barrier),
        // then wait only for tile u (TDM ops complete in order per wave).
        size_t on = (((size_t)b * NT_ + (u + 1)) * KB_) * 32 * 8;
        tdm_load_1d((uint32_t)(uintptr_t)&s_bhi[nxt][0], pBhi + on, TILE_DW);
        tdm_load_1d((uint32_t)(uintptr_t)&s_blo[nxt][0], pBlo + on, TILE_DW);
        __builtin_amdgcn_s_wait_tensorcnt(2);
      } else {
        __builtin_amdgcn_s_wait_tensorcnt(0);
      }
    }
    __syncthreads();   // tile u resident in LDS for everyone

    const uint32_t* lbh = &s_bhi[cur][0];
    const uint32_t* lbl = &s_blo[cur][0];
    v8f acc = {};
#pragma unroll
    for (int kb = 0; kb < KB_; ++kb) {   // K=256: 8 blocks x 3 compensated WMMAs
      int o = (kb * 32 + lane) * 8;
      v16bf bh = *(const v16bf*)(lbh + o);
      v16bf bl = *(const v16bf*)(lbl + o);
      acc = __builtin_amdgcn_wmma_f32_16x16x32_bf16(false, ahi[kb], false, bh,
                                                    (short)0, acc, false, false);
      acc = __builtin_amdgcn_wmma_f32_16x16x32_bf16(false, ahi[kb], false, bl,
                                                    (short)0, acc, false, false);
      acc = __builtin_amdgcn_wmma_f32_16x16x32_bf16(false, alo[kb], false, bh,
                                                    (short)0, acc, false, false);
    }

    // Streaming top-32 update. C/D layout: VGPR v -> row v + 8*half, col = li.
    int m0 = u * 16;
#pragma unroll
    for (int v = 0; v < 8; ++v) {
      int row = v + half * 8;
      int rb  = wave * 16 + row;
      float val = acc[v];
      for (;;) {
        float thr = s_minv[rb];
        unsigned bal = __builtin_amdgcn_ballot_w32(val > thr);
        unsigned mymask = (bal >> (half * 16)) & 0xFFFFu;
        if (mymask == 0) break;              // uniform per 16-lane half
        int src = __ffs(mymask) - 1;
        float cv = lanebcast_f(val, src + half * 16);
        int   ci = m0 + src;
        if (li == src) val = NEG_INF;        // consume this candidate
        int slot = s_mins[rb];
        if (li == 0) { s_vals[rb * TOPK + slot] = cv; s_idxs[rb * TOPK + slot] = ci; }
        // 16-lane cooperative rescan of the 32 slots for the new min
        float v0 = s_vals[rb * TOPK + li];
        float v1 = s_vals[rb * TOPK + li + 16];
        float mv; int ms;
        if (v1 < v0) { mv = v1; ms = li + 16; } else { mv = v0; ms = li; }
#pragma unroll
        for (int off = 8; off > 0; off >>= 1) {  // stays within the half
          float omv = lanebcast_f(mv, lane ^ off);
          int   oms = lanebcast_i(ms, lane ^ off);
          if (omv < mv) { mv = omv; ms = oms; }
        }
        s_minv[rb] = mv; s_mins[rb] = ms;
      }
    }
    __syncthreads();   // everyone done reading buffer `cur` before it is reused
  }

  // Finalize: each wave's list already covers all columns for its 16 rows.
  // Sort to ranks (lower-index tie-break like jax top_k), rescore in exact
  // f32 from the original features, emit all four outputs.
  for (int r = 0; r < 16; ++r) {
    int rb = wave * 16 + r;
    int n  = t * 16 + r;
    float cv = s_vals[rb * TOPK + lane];
    int   ci = s_idxs[rb * TOPK + lane];
    int myIdx = 0;
    for (int j = 0; j < TOPK; ++j) {
      float bv = cv; int bi = ci;
#pragma unroll
      for (int off = 16; off > 0; off >>= 1) {
        float ov = lanebcast_f(bv, lane ^ off);
        int   oi = lanebcast_i(bi, lane ^ off);
        if (ov > bv || (ov == bv && oi < bi)) { bv = ov; bi = oi; }
      }
      if (lane == j) myIdx = bi;             // lane j owns rank j
      if (ci == bi) cv = NEG_INF;            // consume (column indices unique)
    }

    const float* f1p = f1 + (size_t)b * C_ * N_ + n;
    const float* f2p = f2 + (size_t)b * C_ * N_ + myIdx;
    float dot = 0.f;
#pragma unroll 8
    for (int c = 0; c < C_; ++c)
      dot = fmaf(f1p[(size_t)c * N_], f2p[(size_t)c * N_], dot);
    dot *= 0.0625f;  // 1/sqrt(256)

    const size_t KN = (size_t)TOPK * N_;
    float* corr = out;                       // [B,k,N]
    float* c0   = out + (size_t)B_ * KN;     // [B,2,k,N]
    float* c1   = c0 + (size_t)2 * B_ * KN;  // [B,2,k,N]
    float* bx   = c1 + (size_t)2 * B_ * KN;  // [B,1,k,N]
    int j = lane;
    float yn = (float)(n >> 7),     xn = (float)(n & (W_ - 1));
    float ym = (float)(myIdx >> 7), xm = (float)(myIdx & (W_ - 1));
    corr[((size_t)b * TOPK + j) * N_ + n] = dot;
    c0[(((size_t)b * 2 + 0) * TOPK + j) * N_ + n] = yn;
    c0[(((size_t)b * 2 + 1) * TOPK + j) * N_ + n] = xn;
    c1[(((size_t)b * 2 + 0) * TOPK + j) * N_ + n] = ym - yn;
    c1[(((size_t)b * 2 + 1) * TOPK + j) * N_ + n] = xm - xn;
    bx[((size_t)b * TOPK + j) * N_ + n] = (float)b;
  }
}

extern "C" void kernel_launch(void* const* d_in, const int* in_sizes, int n_in,
                              void* d_out, int out_size, void* d_ws, size_t ws_size,
                              hipStream_t stream) {
  (void)in_sizes; (void)n_in; (void)out_size; (void)ws_size;
  const float* f1 = (const float*)d_in[0];
  const float* f2 = (const float*)d_in[1];
  // d_in[2] is k == 32, baked in as TOPK.

  // Workspace: 4 packed tensors (A/B x hi/lo), 8 MB each = 32 MB total.
  size_t seg = (size_t)B_ * NT_ * KB_ * 32 * 8;  // dwords per packed tensor
  uint32_t* pAhi = (uint32_t*)d_ws;
  uint32_t* pAlo = pAhi + seg;
  uint32_t* pBhi = pAlo + seg;
  uint32_t* pBlo = pBhi + seg;

  int packThreads = B_ * NT_ * KB_ * 32;         // 262144
  RAFTGMA_pack<<<packThreads / 256, 256, 0, stream>>>(f1, f2, pAhi, pAlo, pBhi, pBlo);
  RAFTGMA_main<<<B_ * (N_ / 128), 256, 0, stream>>>(f1, f2, pAhi, pAlo, pBhi, pBlo,
                                                    (float*)d_out);
}
